// GraphAttentionLayer_89111981457422
// MI455X (gfx1250) — matrix operand order repaired and checked by reading, not compiled, not run on hardware
//
#include <hip/hip_runtime.h>

typedef __attribute__((ext_vector_type(16))) _Float16 v16h;
typedef __attribute__((ext_vector_type(8)))  float    v8f;

#define N_NODES 8192
#define DIN     512
#define DOUT    64
#define LRA     0.2f   // leaky_relu negative slope

static __device__ __forceinline__ v8f wmma_f16(v16h a, v16h b, v8f c) {
    return __builtin_amdgcn_wmma_f32_16x16x32_f16(
        /*neg_a=*/false, a, /*neg_b=*/false, b,
        /*c_mod=*/(short)0, c, /*reuse_a=*/false, /*reuse_b=*/false);
}

// ---------------------------------------------------------------------------
// K1: h = input @ W   (8192x512 @ 512x64), f16 WMMA, f32 accumulate.
// One wave per 16x16 output tile. 2048 tiles -> 256 blocks x 8 waves.
// ---------------------------------------------------------------------------
__global__ void __launch_bounds__(256)
k_h_gemm(const float* __restrict__ inp, const float* __restrict__ W,
         float* __restrict__ hout) {
    int wid  = (blockIdx.x * blockDim.x + threadIdx.x) >> 5; // global wave id
    int lane = threadIdx.x & 31;
    int tileM = wid >> 2;          // 0..511
    int tileN = wid & 3;           // 0..3
    int m  = lane & 15;
    int h2 = lane >> 4;
    int row  = tileM * 16 + m;
    int ncol = tileN * 16 + m;     // B-frag column for this lane

    v8f c = {};
    for (int k0 = 0; k0 < DIN; k0 += 32) {
        // A fragment: lane holds row `row`, K = k0+16*h2+0..15
        const float* ap = inp + (size_t)row * DIN + k0 + 16 * h2;
        float4 a0 = *(const float4*)(ap);
        float4 a1 = *(const float4*)(ap + 4);
        float4 a2 = *(const float4*)(ap + 8);
        float4 a3 = *(const float4*)(ap + 12);
        float af[16] = {a0.x,a0.y,a0.z,a0.w, a1.x,a1.y,a1.z,a1.w,
                        a2.x,a2.y,a2.z,a2.w, a3.x,a3.y,a3.z,a3.w};
        v16h A;
        #pragma unroll
        for (int t = 0; t < 16; ++t) A[t] = (_Float16)af[t];

        // B fragment: lane holds col `ncol`, K = k0 + 16*h2 + ev (ev = 0..15)
        const float* wp = W + (size_t)(k0 + 16 * h2) * DOUT + ncol;
        v16h B;
        #pragma unroll
        for (int ev = 0; ev < 16; ++ev) B[ev] = (_Float16)wp[(size_t)ev * DOUT];

        c = wmma_f16(A, B, c);
    }
    // C layout: VGPR r -> M = r + 8*h2, N = lane&15
    #pragma unroll
    for (int r = 0; r < 8; ++r)
        hout[(size_t)(tileM * 16 + r + 8 * h2) * DOUT + tileN * 16 + m] = c[r];
}

// ---------------------------------------------------------------------------
// K2: s1[i] = h[i,:]@a1, s2[i] = h[i,:]@a2, per-block max of s2.
// ---------------------------------------------------------------------------
__global__ void __launch_bounds__(256)
k_scores(const float* __restrict__ h, const float* __restrict__ a,
         float* __restrict__ s1, float* __restrict__ s2,
         float* __restrict__ blkmax) {
    int i = blockIdx.x * 256 + threadIdx.x;
    const float* hr = h + (size_t)i * DOUT;
    float v1 = 0.f, v2 = 0.f;
    #pragma unroll 8
    for (int d = 0; d < DOUT; ++d) {
        float hv = hr[d];
        v1 += hv * a[d];
        v2 += hv * a[DOUT + d];
    }
    s1[i] = v1;
    s2[i] = v2;

    __shared__ float red[256];
    red[threadIdx.x] = v2;
    __syncthreads();
    for (int s = 128; s > 0; s >>= 1) {
        if (threadIdx.x < s)
            red[threadIdx.x] = fmaxf(red[threadIdx.x], red[threadIdx.x + s]);
        __syncthreads();
    }
    if (threadIdx.x == 0) blkmax[blockIdx.x] = red[0];
}

// K2b: reduce 32 partial maxima -> s2max
__global__ void k_maxred(const float* __restrict__ blkmax, float* __restrict__ s2max) {
    __shared__ float red[32];
    red[threadIdx.x] = blkmax[threadIdx.x];
    __syncthreads();
    for (int s = 16; s > 0; s >>= 1) {
        if (threadIdx.x < s)
            red[threadIdx.x] = fmaxf(red[threadIdx.x], red[threadIdx.x + s]);
        __syncthreads();
    }
    if (threadIdx.x == 0) *s2max = red[0];
}

// ---------------------------------------------------------------------------
// K3: pre-swizzle h (f16) into WMMA B-fragment order:
//   hswz[((jt*4 + nb)*32 + lane)*16 + ev] = h[jt*32 + 16*(lane>>4) + ev][nb*16 + (lane&15)]
// so the attention kernel loads a whole B fragment per lane contiguously.
// ---------------------------------------------------------------------------
__global__ void __launch_bounds__(256)
k_pack(const float* __restrict__ h, _Float16* __restrict__ hswz) {
    int f    = blockIdx.x * 256 + threadIdx.x;   // 0 .. 8192*64-1
    int ev   = f & 15;
    int lane = (f >> 4) & 31;
    int nb   = (f >> 9) & 3;
    int jt   = f >> 11;
    int j    = jt * 32 + 16 * (lane >> 4) + ev;
    int col  = nb * 16 + (lane & 15);
    hswz[f]  = (_Float16)h[(size_t)j * DOUT + col];
}

// ---------------------------------------------------------------------------
// K4: fused masked-softmax attention @ h, flash-style single pass over adj.
// Block = 16 rows x 8 waves; each wave streams 16x32 adj tiles and issues
// 4x v_wmma_f32_16x16x32_f16 per tile (DOUT=64). Cross-wave combine in LDS.
// ---------------------------------------------------------------------------
__global__ void __launch_bounds__(256)
k_attn(const float* __restrict__ adj, const float* __restrict__ s1,
       const float* __restrict__ s2, const float* __restrict__ s2max_p,
       const v16h* __restrict__ hswz, float* __restrict__ out) {
    __shared__ float numL[16 * DOUT];
    __shared__ float denL[16];
    __shared__ float s1L[16];
    __shared__ float emaxL[16];

    int tid = threadIdx.x;
    int i0  = blockIdx.x * 16;

    #pragma unroll
    for (int q = 0; q < 4; ++q) numL[tid + 256 * q] = 0.f;
    if (tid < 16) {
        denL[tid] = 0.f;
        float v = s1[i0 + tid];
        s1L[tid] = v;
        float s = v + *s2max_p;                       // unmasked row max arg
        emaxL[tid] = (s >= 0.f) ? s : LRA * s;        // leaky_relu monotone
    }
    __syncthreads();

    int wv   = tid >> 5;
    int lane = tid & 31;
    int m    = lane & 15;
    int h2   = lane >> 4;
    float s1m   = s1L[m];
    float emaxm = emaxL[m];

    v8f c0 = {}, c1 = {}, c2 = {}, c3 = {};
    float dsum = 0.f;
    const float* adjrow = adj + (size_t)(i0 + m) * N_NODES;

    for (int it = 0; it < 32; ++it) {
        int j0  = (it * 8 + wv) * 32;
        int kb0 = j0 + 16 * h2;

        float4 d0 = *(const float4*)(adjrow + kb0);
        float4 d1 = *(const float4*)(adjrow + kb0 + 4);
        float4 d2 = *(const float4*)(adjrow + kb0 + 8);
        float4 d3 = *(const float4*)(adjrow + kb0 + 12);
        float4 t0 = *(const float4*)(s2 + kb0);
        float4 t1 = *(const float4*)(s2 + kb0 + 4);
        float4 t2 = *(const float4*)(s2 + kb0 + 8);
        float4 t3 = *(const float4*)(s2 + kb0 + 12);

        float adv[16] = {d0.x,d0.y,d0.z,d0.w, d1.x,d1.y,d1.z,d1.w,
                         d2.x,d2.y,d2.z,d2.w, d3.x,d3.y,d3.z,d3.w};
        float s2v[16] = {t0.x,t0.y,t0.z,t0.w, t1.x,t1.y,t1.z,t1.w,
                         t2.x,t2.y,t2.z,t2.w, t3.x,t3.y,t3.z,t3.w};

        v16h A;
        #pragma unroll
        for (int t = 0; t < 16; ++t) {
            float e = s1m + s2v[t];
            e = (e >= 0.f) ? e : LRA * e;                 // leaky_relu
            float p = (adv[t] > 0.f) ? __expf(e - emaxm) : 0.f;
            dsum += p;
            A[t] = (_Float16)p;
        }

        size_t bb = ((size_t)(j0 >> 5) * 4) * 32 + lane;
        v16h B0 = hswz[bb];
        v16h B1 = hswz[bb + 32];
        v16h B2 = hswz[bb + 64];
        v16h B3 = hswz[bb + 96];

        c0 = wmma_f16(A, B0, c0);
        c1 = wmma_f16(A, B1, c1);
        c2 = wmma_f16(A, B2, c2);
        c3 = wmma_f16(A, B3, c3);
    }

    // combine partial numerators / denominators across waves via LDS atomics
    #pragma unroll
    for (int r = 0; r < 8; ++r) {
        int mm = r + 8 * h2;
        atomicAdd(&numL[mm * DOUT +  0 + m], c0[r]);
        atomicAdd(&numL[mm * DOUT + 16 + m], c1[r]);
        atomicAdd(&numL[mm * DOUT + 32 + m], c2[r]);
        atomicAdd(&numL[mm * DOUT + 48 + m], c3[r]);
    }
    atomicAdd(&denL[m], dsum);
    __syncthreads();

    // out = elu(num/den)
    #pragma unroll
    for (int q = 0; q < 4; ++q) {
        int e   = tid + 256 * q;
        int mm  = e >> 6;
        int col = e & 63;
        float x = numL[e] / denL[mm];
        float o = (x > 0.f) ? x : (__expf(x) - 1.f);
        out[(size_t)(i0 + mm) * DOUT + col] = o;
    }
}

// ---------------------------------------------------------------------------
// Workspace layout (bytes):
//   h_f32 : 0                 (8192*64*4  = 2,097,152)
//   hswz  : 2,097,152         (8192*64*2  = 1,048,576)
//   s1    : 3,145,728         (8192*4)
//   s2    : 3,178,496         (8192*4)
//   blkmax: 3,211,264         (32*4)
//   s2max : 3,211,392         (4)
// ---------------------------------------------------------------------------
extern "C" void kernel_launch(void* const* d_in, const int* in_sizes, int n_in,
                              void* d_out, int out_size, void* d_ws, size_t ws_size,
                              hipStream_t stream) {
    const float* inp = (const float*)d_in[0];   // 8192 x 512
    const float* adj = (const float*)d_in[1];   // 8192 x 8192
    const float* W   = (const float*)d_in[2];   // 512 x 64
    const float* a   = (const float*)d_in[3];   // 128
    float* out       = (float*)d_out;           // 8192 x 64

    char* ws = (char*)d_ws;
    float*     h_f32  = (float*)(ws);
    _Float16*  hswz   = (_Float16*)(ws + 2097152);
    float*     s1     = (float*)(ws + 3145728);
    float*     s2     = (float*)(ws + 3178496);
    float*     blkmax = (float*)(ws + 3211264);
    float*     s2max  = (float*)(ws + 3211392);

    k_h_gemm<<<256, 256, 0, stream>>>(inp, W, h_f32);
    k_scores<<<32, 256, 0, stream>>>(h_f32, a, s1, s2, blkmax);
    k_maxred<<<1, 32, 0, stream>>>(blkmax, s2max);
    k_pack<<<2048, 256, 0, stream>>>(h_f32, hswz);
    k_attn<<<512, 256, 0, stream>>>(adj, s1, s2, s2max,
                                    (const v16h*)hswz, out);
}
